// masked_multihead_attention_73418170958179
// MI455X (gfx1250) — matrix operand order repaired and checked by reading, not compiled, not run on hardware
//
#include <hip/hip_runtime.h>
#include <hip/hip_bf16.h>
#include <math.h>
#include <stdint.h>

typedef _Float16 h8  __attribute__((ext_vector_type(8)));
typedef _Float16 v16h __attribute__((ext_vector_type(16)));
typedef float    v8f  __attribute__((ext_vector_type(8)));

#define WMMA_F32_F16(a, b, c) \
    __builtin_amdgcn_wmma_f32_16x16x32_f16(false, (a), false, (b), (short)0, (c), false, false)

// ---- CDNA5 async global->LDS copy (16B per lane, ASYNCcnt-tracked) --------
// AS3 pointer values are byte offsets in the kernel's LDS block; the flat
// pointer's low 32 bits carry that offset, which is what VDST wants.
__device__ __forceinline__ void async_copy_b128(const _Float16* g, _Float16* l) {
    unsigned lds_off = (unsigned)(uintptr_t)l;
    unsigned long long ga = (unsigned long long)(uintptr_t)g;
    asm volatile("global_load_async_to_lds_b128 %0, %1, off"
                 :: "v"(lds_off), "v"(ga) : "memory");
}
__device__ __forceinline__ void async_wait_all() {
    asm volatile("s_wait_asynccnt 0x0" ::: "memory");
}

// ---- fragment loaders (wave32, 16x16x32 f16 WMMA) -------------------------
// A 16x32 f16: lane L holds row M=L%16; lo lanes K={0..7,16..23}, hi lanes +8.
__device__ __forceinline__ v16h frag_a(const _Float16* p) {
    h8 lo = *(const h8*)(p);
    h8 hi = *(const h8*)(p + 16);
    return __builtin_shufflevector(lo, hi, 0,1,2,3,4,5,6,7,8,9,10,11,12,13,14,15);
}
// B 32x16 f16 staged K-major ([N][K]): lane L holds col N=L%16,
// lo lanes K={0..7,16..23}, hi lanes +8.
__device__ __forceinline__ v16h frag_b(const _Float16* p) {
    h8 lo = *(const h8*)(p);
    h8 hi = *(const h8*)(p + 16);
    return __builtin_shufflevector(lo, hi, 0,1,2,3,4,5,6,7,8,9,10,11,12,13,14,15);
}

// ---- 1) f32 -> f16 convert ------------------------------------------------
__global__ __launch_bounds__(256) void cvt_f32_f16(const float* __restrict__ in,
                                                   _Float16* __restrict__ out) {
    int i = (blockIdx.x * 256 + threadIdx.x) * 8;
    float4 a = *(const float4*)(in + i);
    float4 b = *(const float4*)(in + i + 4);
    h8 o;
    o[0]=(_Float16)a.x; o[1]=(_Float16)a.y; o[2]=(_Float16)a.z; o[3]=(_Float16)a.w;
    o[4]=(_Float16)b.x; o[5]=(_Float16)b.y; o[6]=(_Float16)b.z; o[7]=(_Float16)b.w;
    *(h8*)(out + i) = o;
}

#define LDA 40   // halves; 80B row stride, keeps 16B-aligned fragment loads
#define LDB 40

// ---- 2) C_f16[M,N] = A_f16[M,K] @ B_f16[K,N] + bias_f32[N] ---------------
// Block tile 128x64, BK=32, 8 waves as 4x2, each wave 2x2 WMMA tiles.
// A tile staged with GLOBAL_LOAD_ASYNC_TO_LDS_B128 (no VGPR round-trip).
__global__ __launch_bounds__(256) void gemm_bias_f16(
    const _Float16* __restrict__ A, const _Float16* __restrict__ Bm,
    const float* __restrict__ bias, _Float16* __restrict__ C,
    int M, int N, int K) {
    __shared__ __attribute__((aligned(16))) _Float16 sA[128 * LDA];
    __shared__ __attribute__((aligned(16))) _Float16 sB[64 * LDB];
    const int tid = threadIdx.x, lane = tid & 31, wave = tid >> 5;
    const int wm = wave >> 1, wn = wave & 1;
    const int m0 = blockIdx.y * 128, n0 = blockIdx.x * 64;
    v8f acc[2][2] = {};
    for (int k0 = 0; k0 < K; k0 += 32) {
        #pragma unroll
        for (int i = 0; i < 2; ++i) {            // A: 128x32 halves, async DMA
            int lin = tid + i * 256;
            int row = lin >> 2, c = (lin & 3) << 3;
            async_copy_b128(&A[(size_t)(m0 + row) * K + k0 + c], &sA[row * LDA + c]);
        }
        #pragma unroll
        for (int i = 0; i < 8; ++i) {            // B: 32x64 -> sB[n][k]
            int lin = tid + i * 256;
            int kk = lin >> 6, nn = lin & 63;
            sB[nn * LDB + kk] = Bm[(size_t)(k0 + kk) * N + n0 + nn];
        }
        async_wait_all();
        __syncthreads();
        v16h af[2], bf[2];
        #pragma unroll
        for (int i = 0; i < 2; ++i)
            af[i] = frag_a(&sA[(wm * 32 + i * 16 + (lane & 15)) * LDA + ((lane >> 4) << 3)]);
        #pragma unroll
        for (int j = 0; j < 2; ++j)
            bf[j] = frag_b(&sB[(wn * 32 + j * 16 + (lane & 15)) * LDB + ((lane >> 4) << 3)]);
        #pragma unroll
        for (int i = 0; i < 2; ++i)
            #pragma unroll
            for (int j = 0; j < 2; ++j)
                acc[i][j] = WMMA_F32_F16(af[i], bf[j], acc[i][j]);
        __syncthreads();
    }
    const int rh = (lane >> 4) << 3, col = lane & 15;
    #pragma unroll
    for (int i = 0; i < 2; ++i)
        #pragma unroll
        for (int j = 0; j < 2; ++j)
            #pragma unroll
            for (int r = 0; r < 8; ++r) {
                int m = m0 + wm * 32 + i * 16 + rh + r;
                int n = n0 + wn * 32 + j * 16 + col;
                C[(size_t)m * N + n] = (_Float16)(acc[i][j][r] + bias[n]);
            }
}

// ---- 3) raw scores: S[bh,q,k] = (Q . K) * NU^-0.5, skip fully-masked -----
__global__ __launch_bounds__(256) void scores_qk(
    const _Float16* __restrict__ Q, const _Float16* __restrict__ Kp,
    float* __restrict__ Wt) {
    const int T = 2048, NUc = 1024, DH = 64;
    const int bh = blockIdx.z, b = bh >> 4, h = bh & 15;
    const int qb = blockIdx.y * 64, kb = blockIdx.x * 32;
    if (kb > qb + 63) return;                    // whole block above diagonal
    const int lane = threadIdx.x & 31, wave = threadIdx.x >> 5;
    const int q0 = qb + (wave >> 1) * 16, k0 = kb + (wave & 1) * 16;
    const _Float16* Qb = Q  + ((size_t)b * T + q0) * NUc + h * DH;
    const _Float16* Kb = Kp + ((size_t)b * T + k0) * NUc + h * DH;
    v8f acc = {};
    #pragma unroll
    for (int d0 = 0; d0 < DH; d0 += 32) {
        // A row (q) and B col (key row) are both contiguous along d in memory.
        v16h a  = frag_a(Qb + d0 + (size_t)(lane & 15) * NUc + ((lane >> 4) << 3));
        v16h bb = frag_b(Kb + d0 + (size_t)(lane & 15) * NUc + ((lane >> 4) << 3));
        acc = WMMA_F32_F16(a, bb, acc);
    }
    float* out = Wt + (size_t)bh * T * T;
    const int rh = (lane >> 4) << 3, col = lane & 15;
    #pragma unroll
    for (int r = 0; r < 8; ++r)
        out[(size_t)(q0 + rh + r) * T + k0 + col] = acc[r] * 0.03125f; // 1024^-0.5
}

// ---- 4) causal softmax in place over each row of weights -----------------
__global__ __launch_bounds__(256) void softmax_causal(float* __restrict__ Wt) {
    const int T = 2048;
    __shared__ float red[256];
    const size_t row = blockIdx.x;
    const int q = (int)(row & (T - 1));
    float* p = Wt + row * (size_t)T;
    const int tid = threadIdx.x, nv = q + 1;
    float m = -INFINITY;
    for (int k = tid; k < nv; k += 256) m = fmaxf(m, p[k]);
    red[tid] = m; __syncthreads();
    for (int s = 128; s > 0; s >>= 1) {
        if (tid < s) red[tid] = fmaxf(red[tid], red[tid + s]);
        __syncthreads();
    }
    m = red[0]; __syncthreads();
    float sum = 0.f;
    for (int k = tid; k < nv; k += 256) sum += __expf(p[k] - m);
    red[tid] = sum; __syncthreads();
    for (int s = 128; s > 0; s >>= 1) {
        if (tid < s) red[tid] += red[tid + s];
        __syncthreads();
    }
    const float inv = 1.f / red[0];
    for (int k = tid; k < T; k += 256)
        p[k] = (k < nv) ? __expf(p[k] - m) * inv : 0.f;
}

// ---- 5) out[b,q,h*64+d] = sum_k P[bh,q,k] * V[b,k,h*64+d] ----------------
// Block tile 128q x 64d, BK=32, causal K-loop truncation. P cvt'd f32->f16.
__global__ __launch_bounds__(256) void av_gemm(
    const float* __restrict__ Wt, const _Float16* __restrict__ V,
    float* __restrict__ Out) {
    const int T = 2048, NUc = 1024, DH = 64;
    __shared__ __attribute__((aligned(16))) _Float16 sA[128 * LDA];
    __shared__ __attribute__((aligned(16))) _Float16 sB[64 * LDB];
    const int bh = blockIdx.y, b = bh >> 4, h = bh & 15;
    const int q0 = blockIdx.x * 128;
    const int tid = threadIdx.x, lane = tid & 31, wave = tid >> 5;
    const int wm = wave >> 1, wn = wave & 1;
    const float* Wp = Wt + (size_t)bh * T * T;
    v8f acc[2][2] = {};
    const int kend = q0 + 128;                   // probs are 0 beyond diagonal
    for (int k0 = 0; k0 < kend; k0 += 32) {
        #pragma unroll
        for (int i = 0; i < 2; ++i) {            // P tile 128x32 f32 -> f16 LDS
            int lin = tid + i * 256;
            int row = lin >> 2, c = (lin & 3) << 3;
            const float* src = Wp + (size_t)(q0 + row) * T + k0 + c;
            if (k0 + 32 < kend)                  // pull next K-step into cache
                __builtin_prefetch(src + 32, 0, 0);
            h8 o;
            #pragma unroll
            for (int jj = 0; jj < 8; ++jj) o[jj] = (_Float16)src[jj];
            *(h8*)&sA[row * LDA + c] = o;
        }
        #pragma unroll
        for (int i = 0; i < 8; ++i) {            // V tile 32x64 -> sB[d][k]
            int lin = tid + i * 256;
            int kk = lin >> 6, dd = lin & 63;
            sB[dd * LDB + kk] = V[((size_t)b * T + k0 + kk) * NUc + h * DH + dd];
        }
        __syncthreads();
        v16h af[2], bf[2];
        #pragma unroll
        for (int i = 0; i < 2; ++i)
            af[i] = frag_a(&sA[(wm * 32 + i * 16 + (lane & 15)) * LDA + ((lane >> 4) << 3)]);
        #pragma unroll
        for (int j = 0; j < 2; ++j)
            bf[j] = frag_b(&sB[(wn * 32 + j * 16 + (lane & 15)) * LDB + ((lane >> 4) << 3)]);
        #pragma unroll
        for (int i = 0; i < 2; ++i)
            #pragma unroll
            for (int j = 0; j < 2; ++j)
                acc[i][j] = WMMA_F32_F16(af[i], bf[j], acc[i][j]);
        __syncthreads();
    }
    const int rh = (lane >> 4) << 3, col = lane & 15;
    #pragma unroll
    for (int i = 0; i < 2; ++i)
        #pragma unroll
        for (int j = 0; j < 2; ++j)
            #pragma unroll
            for (int r = 0; r < 8; ++r)
                Out[((size_t)b * T + q0 + wm * 32 + i * 16 + rh + r) * NUc
                    + h * DH + wn * 32 + j * 16 + col] = acc[i][j][r];
}

extern "C" void kernel_launch(void* const* d_in, const int* in_sizes, int n_in,
                              void* d_out, int out_size, void* d_ws, size_t ws_size,
                              hipStream_t stream) {
    const int Bc = 2, Tc = 2048, Dc = 1024, Hc = 16, NUc = 1024;
    const float* queries = (const float*)d_in[0];
    const float* keys    = (const float*)d_in[1];
    const float* Wq = (const float*)d_in[2];
    const float* bq = (const float*)d_in[3];
    const float* Wk = (const float*)d_in[4];
    const float* bk = (const float*)d_in[5];
    const float* Wv = (const float*)d_in[6];
    const float* bv = (const float*)d_in[7];

    float* out     = (float*)d_out;                       // [B,T,NU]
    float* weights = out + (size_t)Bc * Tc * NUc;         // [B,H,T,T]

    const size_t nAct = (size_t)Bc * Tc * Dc;             // 4194304
    const size_t nW   = (size_t)Dc * NUc;                 // 1048576
    _Float16* qh  = (_Float16*)d_ws;
    _Float16* kh  = qh  + nAct;
    _Float16* wqh = kh  + nAct;
    _Float16* wkh = wqh + nW;
    _Float16* wvh = wkh + nW;
    _Float16* Qp  = wvh + nW;
    _Float16* Kp  = Qp  + nAct;
    _Float16* Vp  = Kp  + nAct;

    cvt_f32_f16<<<(unsigned)(nAct / 2048), 256, 0, stream>>>(queries, qh);
    cvt_f32_f16<<<(unsigned)(nAct / 2048), 256, 0, stream>>>(keys,    kh);
    cvt_f32_f16<<<(unsigned)(nW   / 2048), 256, 0, stream>>>(Wq, wqh);
    cvt_f32_f16<<<(unsigned)(nW   / 2048), 256, 0, stream>>>(Wk, wkh);
    cvt_f32_f16<<<(unsigned)(nW   / 2048), 256, 0, stream>>>(Wv, wvh);

    dim3 g1(NUc / 64, (Bc * Tc) / 128);
    gemm_bias_f16<<<g1, 256, 0, stream>>>(qh, wqh, bq, Qp, Bc * Tc, NUc, Dc);
    gemm_bias_f16<<<g1, 256, 0, stream>>>(kh, wkh, bk, Kp, Bc * Tc, NUc, Dc);
    gemm_bias_f16<<<g1, 256, 0, stream>>>(kh, wvh, bv, Vp, Bc * Tc, NUc, Dc);

    dim3 g2(Tc / 32, Tc / 64, Bc * Hc);
    scores_qk<<<g2, 256, 0, stream>>>(Qp, Kp, weights);

    softmax_causal<<<(unsigned)(Bc * Hc * Tc), 256, 0, stream>>>(weights);

    dim3 g3(Tc / 128, Bc * Hc);
    av_gemm<<<g3, 256, 0, stream>>>(weights, Vp, out);
}
